// DeformBlock_38757784879812
// MI455X (gfx1250) — matrix operand order repaired and checked by reading, not compiled, not run on hardware
//
#include <hip/hip_runtime.h>
#include <hip/hip_bf16.h>

// ---------------------------------------------------------------------------
// Deformable 3D conv block for MI455X (gfx1250, wave32, WMMA).
// Both GEMM stages run on v_wmma_f32_16x16x32_f16 (f16 in, f32 accumulate).
// Padded activation is kept channel-contiguous f16 so trilinear corner
// gathers are single contiguous 16B loads and implicit im2col maps one
// 3x3x3 tap (32 channels) onto exactly one WMMA K=32 slice.
// Working set (~34 MB) is L2-resident (192 MB); HBM traffic ~20 MB.
// ---------------------------------------------------------------------------

typedef __attribute__((ext_vector_type(16))) _Float16 v16h;
typedef __attribute__((ext_vector_type(8)))  _Float16 v8h;
typedef __attribute__((ext_vector_type(8)))  float    v8f;

#define CIN   32
#define COUT  32
#define DDIM  32
#define HDIM  48
#define WDIM  48
#define DP    34
#define HP    50
#define WP    50
#define NTAP  27
#define KTOT  864      // 27 taps * 32 channels
#define NOPAD 96       // 81 offset channels padded to 96 (6 N-tiles)

union FragH { v16h v; v8h h[2]; };

// workspace byte offsets (all 256B aligned)
#define WS_OFFS 0u                         // 73728*96 f32  = 28,311,552 B
#define WS_XP   28311552u                  // 34*50*50*32 f16 = 5,440,000 B
#define WS_WH   (28311552u + 5440000u)     // 96*864 f16     =   165,888 B
#define WS_WD   (28311552u + 5440000u + 165888u) // 32*864 f16 =  55,296 B

// ---------------- pack x -> padded [dp][hp][wp][c] f16 ----------------------
__global__ void pack_x(const float* __restrict__ x, _Float16* __restrict__ xp) {
    int v = blockIdx.x * blockDim.x + threadIdx.x;
    if (v >= DP * HP * WP) return;
    int wp = v % WP;
    int hp = (v / WP) % HP;
    int dp = v / (WP * HP);
    _Float16* dst = xp + (size_t)v * CIN;
    int d = dp - 1, h = hp - 1, w = wp - 1;
    if ((unsigned)d < DDIM && (unsigned)h < HDIM && (unsigned)w < WDIM) {
        const float* src = x + ((size_t)d * HDIM + h) * WDIM + w;
        #pragma unroll
        for (int c = 0; c < CIN; ++c)
            dst[c] = (_Float16)src[(size_t)c * (DDIM * HDIM * WDIM)];
    } else {
        #pragma unroll
        for (int c = 0; c < CIN; ++c) dst[c] = (_Float16)0.f;
    }
}

// ---------------- pack weights into B-matrix layout [o][n*32+c] -------------
__global__ void pack_w(const float* __restrict__ w_off, const float* __restrict__ w_def,
                       _Float16* __restrict__ Wh, _Float16* __restrict__ Wd) {
    int idx = blockIdx.x * blockDim.x + threadIdx.x;
    const int NH = NOPAD * KTOT;   // 82944
    const int ND = COUT * KTOT;    // 27648
    if (idx < NH) {
        int o = idx / KTOT, k = idx % KTOT;
        int n = k / CIN,    c = k % CIN;
        float val = (o < 81) ? w_off[(size_t)(o * CIN + c) * NTAP + n] : 0.f;
        Wh[idx] = (_Float16)val;
    } else if (idx < NH + ND) {
        int j = idx - NH;
        int o = j / KTOT, k = j % KTOT;
        int n = k / CIN,  c = k % CIN;
        Wd[j] = (_Float16)w_def[(size_t)(o * CIN + c) * NTAP + n];
    }
}

// ---------------- offset conv: implicit-im2col WMMA GEMM -------------------
// One workgroup per (d,h): M = 48 positions (3 tiles), N = 96 (6 waves),
// K = 27 taps x 32 channels. x slab staged in LDS once.
__global__ __launch_bounds__(192) void offset_conv(const _Float16* __restrict__ xp,
                                                   const _Float16* __restrict__ Wh,
                                                   float* __restrict__ offs) {
    __shared__ __align__(16) _Float16 ldsX[9 * WP * CIN];   // 28800 B
    int g = blockIdx.x;
    int h = g % HDIM, d = g / HDIM;
    int tid = threadIdx.x;

    // stage 9 contiguous (kd,kh) slabs of [50][32] f16 as dword copies
    const unsigned* xu = (const unsigned*)xp;
    unsigned* lu = (unsigned*)ldsX;
    const int SLICE_DW = WP * CIN / 2;   // 800 dwords
    for (int i = tid; i < 9 * SLICE_DW; i += 192) {
        int r   = i / SLICE_DW;
        int rem = i % SLICE_DW;
        int kd = r / 3, kh = r % 3;
        lu[i] = xu[((size_t)(d + kd) * HP + (h + kh)) * SLICE_DW + rem];
    }
    __syncthreads();

    int lane = tid & 31;
    int nt   = tid >> 5;       // N-tile 0..5
    int row  = lane & 15;
    int hi   = lane >> 4;
    int c0   = hi * 8;         // A-fragment K sub-base (16-bit A layout)
    int kb2  = hi * 8;         // B-fragment K sub-base (16-bit B layout)
    int col  = nt * 16 + row;

    v8f acc0 = {}, acc1 = {}, acc2 = {};
    const _Float16* wbase = Wh + (size_t)col * KTOT + kb2;

    for (int n = 0; n < NTAP; ++n) {
        int kd = n / 9, kh = (n / 3) % 3, kw = n % 3;
        FragH b;
        b.h[0] = *(const v8h*)(wbase + n * CIN);
        b.h[1] = *(const v8h*)(wbase + n * CIN + 16);
        int sbase = (kd * 3 + kh) * (WP * CIN) + kw * CIN + c0;
        FragH a;
        a.h[0] = *(const v8h*)(ldsX + sbase + (0 * 16 + row) * CIN);
        a.h[1] = *(const v8h*)(ldsX + sbase + (0 * 16 + row) * CIN + 16);
        acc0 = __builtin_amdgcn_wmma_f32_16x16x32_f16(false, a.v, false, b.v,
                                                      (short)0, acc0, false, false);
        a.h[0] = *(const v8h*)(ldsX + sbase + (1 * 16 + row) * CIN);
        a.h[1] = *(const v8h*)(ldsX + sbase + (1 * 16 + row) * CIN + 16);
        acc1 = __builtin_amdgcn_wmma_f32_16x16x32_f16(false, a.v, false, b.v,
                                                      (short)0, acc1, false, false);
        a.h[0] = *(const v8h*)(ldsX + sbase + (2 * 16 + row) * CIN);
        a.h[1] = *(const v8h*)(ldsX + sbase + (2 * 16 + row) * CIN + 16);
        acc2 = __builtin_amdgcn_wmma_f32_16x16x32_f16(false, a.v, false, b.v,
                                                      (short)0, acc2, false, false);
    }

    // scatter D fragments: lane r holds (M = r + 8*hi, N = col)
    size_t pb = ((size_t)d * HDIM + h) * WDIM;
    #pragma unroll
    for (int r = 0; r < 8; ++r) {
        int m = r + hi * 8;
        offs[(pb + 0 * 16 + m) * NOPAD + col] = acc0[r];
        offs[(pb + 1 * 16 + m) * NOPAD + col] = acc1[r];
        offs[(pb + 2 * 16 + m) * NOPAD + col] = acc2[r];
    }
}

// ---------------- fused trilinear sampling + deformable GEMM ---------------
// One workgroup per 32-position tile of a d-slice (2304 = 72 x 32 exact).
// 128 threads: 4 threads/position blend one tap's 32 channels into an LDS
// 32x32 f16 A-tile; 4 waves = 2 M-tiles x 2 Cout N-tiles, one WMMA each per
// tap. Ping-pong A-tiles give a single barrier per tap, so sampling of tap
// n+1 overlaps the WMMA of tap n.
__global__ __launch_bounds__(128) void deform_gemm(const _Float16* __restrict__ xp,
                                                   const float* __restrict__ offs,
                                                   const _Float16* __restrict__ Wd,
                                                   float* __restrict__ out) {
    __shared__ __align__(16) _Float16 tile[2][32 * CIN];   // 4 KB ping-pong
    int g = blockIdx.x;            // 0..2303
    int d = g / 72;
    int t = g % 72;                // 32-position tile within the d-slice
    int tid = threadIdx.x;

    int pos  = tid >> 2;           // 0..31
    int sub  = tid & 3;            // channel group 0..3
    int cg   = sub * 8;
    int flat = t * 32 + pos;       // flat (h,w) index, 0..2303
    int h    = flat / WDIM;
    int w    = flat % WDIM;
    const float* offp = offs + (((size_t)d * HDIM + h) * WDIM + w) * NOPAD;
    float p0d = (float)(d + 1), p0h = (float)(h + 1), p0w = (float)(w + 1);

    int lane = tid & 31;
    int wave = tid >> 5;           // 0..3
    int mt   = wave >> 1;          // M-tile 0/1 (positions mt*16..mt*16+15)
    int ntl  = wave & 1;           // Cout N-tile 0/1
    int rowA = lane & 15;
    int hi   = lane >> 4;
    int c0   = hi * 8;
    int kb2  = hi * 8;
    int col  = ntl * 16 + rowA;
    const _Float16* wbase = Wd + (size_t)col * KTOT + kb2;

    v8f acc = {};
    for (int n = 0; n < NTAP; ++n) {
        int kd = n / 9, kh = (n / 3) % 3, kw = n % 3;
        float od = offp[n], oh = offp[27 + n], ow = offp[54 + n];
        float pd = p0d + (float)(kd - 1) + od;
        float ph = p0h + (float)(kh - 1) + oh;
        float pw = p0w + (float)(kw - 1) + ow;
        float fd = floorf(pd), fh = floorf(ph), fw = floorf(pw);
        float qd0 = fminf(fmaxf(fd,        0.f), (float)(DP - 1));
        float qd1 = fminf(fmaxf(fd + 1.f,  0.f), (float)(DP - 1));
        float qh0 = fminf(fmaxf(fh,        0.f), (float)(HP - 1));
        float qh1 = fminf(fmaxf(fh + 1.f,  0.f), (float)(HP - 1));
        float qw0 = fminf(fmaxf(fw,        0.f), (float)(WP - 1));
        float qw1 = fminf(fmaxf(fw + 1.f,  0.f), (float)(WP - 1));
        float pdc = fminf(fmaxf(pd, 0.f), (float)(DP - 1));
        float phc = fminf(fmaxf(ph, 0.f), (float)(HP - 1));
        float pwc = fminf(fmaxf(pw, 0.f), (float)(WP - 1));
        float gdv[2] = {1.f + (qd0 - pdc), 1.f - (qd1 - pdc)};
        float ghv[2] = {1.f + (qh0 - phc), 1.f - (qh1 - phc)};
        float gwv[2] = {1.f + (qw0 - pwc), 1.f - (qw1 - pwc)};
        int   idv[2] = {(int)qd0, (int)qd1};
        int   ihv[2] = {(int)qh0, (int)qh1};
        int   iwv[2] = {(int)qw0, (int)qw1};

        float av[8] = {0.f, 0.f, 0.f, 0.f, 0.f, 0.f, 0.f, 0.f};
        #pragma unroll
        for (int ia = 0; ia < 2; ++ia)
            #pragma unroll
            for (int ib = 0; ib < 2; ++ib)
                #pragma unroll
                for (int ic = 0; ic < 2; ++ic) {
                    float wgt = gdv[ia] * ghv[ib] * gwv[ic];
                    const v8h* src = (const v8h*)(xp +
                        (((size_t)idv[ia] * HP + ihv[ib]) * WP + iwv[ic]) * CIN + cg);
                    v8h xv = *src;   // contiguous 16B corner gather
                    #pragma unroll
                    for (int j = 0; j < 8; ++j) av[j] += wgt * (float)xv[j];
                }
        _Float16* tb = tile[n & 1];
        #pragma unroll
        for (int j = 0; j < 8; ++j) tb[pos * CIN + cg + j] = (_Float16)av[j];
        __syncthreads();   // single barrier per tap (ping-pong buffers)

        FragH a_, b_;
        a_.h[0] = *(const v8h*)(tb + (mt * 16 + rowA) * CIN + c0);
        a_.h[1] = *(const v8h*)(tb + (mt * 16 + rowA) * CIN + c0 + 16);
        b_.h[0] = *(const v8h*)(wbase + n * CIN);
        b_.h[1] = *(const v8h*)(wbase + n * CIN + 16);
        acc = __builtin_amdgcn_wmma_f32_16x16x32_f16(false, a_.v, false, b_.v,
                                                     (short)0, acc, false, false);
    }

    // out is NCDHW f32: lane r holds (flat pos = t*32 + mt*16 + r + 8*hi, o = col)
    #pragma unroll
    for (int r = 0; r < 8; ++r) {
        int fl = t * 32 + mt * 16 + r + hi * 8;
        out[((size_t)col * DDIM + d) * (HDIM * WDIM) + fl] = acc[r];
    }
}

extern "C" void kernel_launch(void* const* d_in, const int* in_sizes, int n_in,
                              void* d_out, int out_size, void* d_ws, size_t ws_size,
                              hipStream_t stream) {
    const float* x     = (const float*)d_in[0];
    const float* w_off = (const float*)d_in[1];
    const float* w_def = (const float*)d_in[2];
    float* out = (float*)d_out;
    char* ws = (char*)d_ws;

    float*    offs = (float*)(ws + WS_OFFS);
    _Float16* xp   = (_Float16*)(ws + WS_XP);
    _Float16* Wh   = (_Float16*)(ws + WS_WH);
    _Float16* Wd   = (_Float16*)(ws + WS_WD);

    (void)in_sizes; (void)n_in; (void)out_size; (void)ws_size;

    const int nvox = DP * HP * WP;                  // 85000 padded voxels
    pack_x<<<(nvox + 255) / 256, 256, 0, stream>>>(x, xp);
    const int nwel = NOPAD * KTOT + COUT * KTOT;    // 110592 weight elements
    pack_w<<<(nwel + 255) / 256, 256, 0, stream>>>(w_off, w_def, Wh, Wd);
    offset_conv<<<DDIM * HDIM, 192, 0, stream>>>(xp, Wh, offs);
    deform_gemm<<<DDIM * HDIM * 3 / 2, 128, 0, stream>>>(xp, offs, Wd, out);
}
